// MyMode_9320079033206
// MI455X (gfx1250) — matrix operand order repaired and checked by reading, not compiled, not run on hardware
//
#include <hip/hip_runtime.h>
#include <math.h>

typedef __attribute__((ext_vector_type(16))) _Float16 v16h;
typedef __attribute__((ext_vector_type(8)))  _Float16 v8h;
typedef __attribute__((ext_vector_type(8)))  float    v8f;
typedef __attribute__((ext_vector_type(4)))  float    v4f;

#define VOCAB  10000
#define EMB    16
#define HID    32
#define BATCH  4096
#define TSTEPS 512

// Hardware tanh: CDNA5 V_TANH_F32 (TRANS32). Avoid the branchy libm expansion.
__device__ __forceinline__ float fast_tanh(float x) {
#if __has_builtin(__builtin_amdgcn_tanhf)
    return __builtin_amdgcn_tanhf(x);
#else
    float y;
    // v_nop covers the TRANS-result hazard (1 independent op before use);
    // the compiler cannot schedule around opaque inline asm.
    asm volatile("v_tanh_f32 %0, %1\n\tv_nop" : "=v"(y) : "v"(x));
    return y;
#endif
}

// One wave32 per 16 batch rows. Recurrence h_t = tanh(x_t@Wx + h_{t-1}@Wh + b)
// via v_wmma_f32_16x16x32_f16 (K=32 == HID exactly; EMB=16 zero-padded).
__global__ __launch_bounds__(32)
void rnn_wmma_kernel(const int*   __restrict__ x,    // [B,T]
                     const float* __restrict__ emb,  // [VOCAB,EMB]
                     const float* __restrict__ Wx,   // [EMB,HID]
                     const float* __restrict__ Wh,   // [HID,HID]
                     const float* __restrict__ bias, // [HID]
                     const float* __restrict__ Wd,   // [HID,1]
                     const float* __restrict__ bd,   // [1]
                     float*       __restrict__ out)  // [B,1]
{
    const int  lane  = threadIdx.x;        // 0..31
    const int  n     = lane & 15;          // N column within a 16-wide tile
    const bool hi    = lane >= 16;
    const int  kbase = hi ? 16 : 0;        // unused; B-matrix K index also uses koff
    const int  koff  = hi ? 8  : 0;        // A-matrix K sub-offset for this lane
    const int  mbase = hi ? 8  : 0;        // C/D-matrix M offset for this lane
    const int  base  = blockIdx.x * 16;    // batch rows [base, base+16)

    __shared__ alignas(16) _Float16 hbuf[16 * HID];   // h tile, row-major f16

    // ---- loop-invariant B matrices in WMMA B-layout (lane holds col n, K=koff..koff+7, 16+koff..16+koff+7)
    v16h Bwh0, Bwh1, Bwx0, Bwx1;
    #pragma unroll
    for (int i = 0; i < 16; ++i) {
        const int k = (i < 8) ? (koff + i) : (16 + koff + (i - 8));
        Bwh0[i] = (_Float16)Wh[k * HID + n];
        Bwh1[i] = (_Float16)Wh[k * HID + n + 16];
        const float wx0 = (k < EMB) ? Wx[k * HID + n]      : 0.0f;  // zero-pad K>=EMB
        const float wx1 = (k < EMB) ? Wx[k * HID + n + 16] : 0.0f;
        Bwx0[i] = (_Float16)wx0;
        Bwx1[i] = (_Float16)wx1;
    }
    const float b0 = bias[n];
    const float b1 = bias[n + 16];

    // ---- h_0 = 0
    #pragma unroll
    for (int i = 0; i < 16; ++i) hbuf[lane * 16 + i] = (_Float16)0.0f;
    asm volatile("s_wait_dscnt 0" ::: "memory");

    const int  m    = n;                                  // A-matrix row for this lane
    const int* xrow = x + (size_t)(base + m) * TSTEPS;

    // ---- 2-deep software pipeline: eA holds emb row for step t,
    //      idxB holds the index for step t+1.
    int idxB = xrow[0];
    {
        // stage-0 fill: load e(t=0), then advance idxB to t=1
        const v4f* ep = (const v4f*)(emb + (size_t)idxB * EMB + koff);
        (void)ep;
    }
    v4f eA0, eA1;
    {
        const v4f* ep = (const v4f*)(emb + (size_t)idxB * EMB + koff); // t=0 row
        eA0 = ep[0];
        eA1 = ep[1];
    }
    idxB = xrow[1 < TSTEPS ? 1 : 0];                      // index for t+1

    for (int t = 0; t < TSTEPS; ++t) {
        // ---- issue prefetch for step t+1 (emb row) and t+2 (index) FIRST,
        //      so global latency overlaps the WMMA/tanh/LDS work below.
        const v4f* epn = (const v4f*)(emb + (size_t)idxB * EMB + koff);
        const v4f  nE0 = epn[0];
        const v4f  nE1 = epn[1];
        const int  tn2 = (t + 2 < TSTEPS) ? (t + 2) : (TSTEPS - 1);
        const int  idxC = xrow[tn2];

        // ---- A_x from the pipelined registers (f32 -> f16, K 16..31 zero-padded)
        v16h Ax;
        #pragma unroll
        for (int i = 0; i < 4; ++i) {
            Ax[i]      = (_Float16)eA0[i];  // K = koff..koff+3
            Ax[4 + i]  = (_Float16)eA1[i];  // K = koff+4..koff+7
            Ax[8 + i]  = (_Float16)0.0f;    // K = 16+koff..  (zero pad)
            Ax[12 + i] = (_Float16)0.0f;
        }

        // ---- A_h: previous h from LDS (row m, K halves {koff..+7, 16+koff..+7})
        const v8h* hp = (const v8h*)(hbuf + m * HID + koff);  // 16B aligned
        const v8h  hl = hp[0];
        const v8h  hh = hp[2];                                // +16 halves
        v16h Ah;
        #pragma unroll
        for (int i = 0; i < 8; ++i) { Ah[i] = hl[i]; Ah[8 + i] = hh[i]; }

        // ---- C = bias; D = A_x*Wx + A_h*Wh + C
        v8f acc0, acc1;
        #pragma unroll
        for (int r = 0; r < 8; ++r) { acc0[r] = b0; acc1[r] = b1; }

        acc0 = __builtin_amdgcn_wmma_f32_16x16x32_f16(false, Ax, false, Bwx0,
                                                      (short)0, acc0, false, false);
        acc1 = __builtin_amdgcn_wmma_f32_16x16x32_f16(false, Ax, false, Bwx1,
                                                      (short)0, acc1, false, false);
        acc0 = __builtin_amdgcn_wmma_f32_16x16x32_f16(false, Ah, false, Bwh0,
                                                      (short)0, acc0, false, false);
        acc1 = __builtin_amdgcn_wmma_f32_16x16x32_f16(false, Ah, false, Bwh1,
                                                      (short)0, acc1, false, false);

        // ---- hardware tanh + write h_t back to LDS (D layout: M = mbase+r)
        #pragma unroll
        for (int r = 0; r < 8; ++r) {
            const int M = mbase + r;
            hbuf[M * HID + n]      = (_Float16)fast_tanh(acc0[r]);
            hbuf[M * HID + n + 16] = (_Float16)fast_tanh(acc1[r]);
        }
        asm volatile("s_wait_dscnt 0" ::: "memory");  // make h_t visible wave-wide

        // ---- rotate pipeline registers
        eA0  = nE0;
        eA1  = nE1;
        idxB = idxC;
    }

    // ---- dense head: out = sigmoid(h_T @ Wd + bd), one row per lane 0..15
    if (lane < 16) {
        float z = bd[0];
        #pragma unroll
        for (int k = 0; k < HID; ++k)
            z += (float)hbuf[lane * HID + k] * Wd[k];
        out[base + lane] = 1.0f / (1.0f + expf(-z));
    }
}

extern "C" void kernel_launch(void* const* d_in, const int* in_sizes, int n_in,
                              void* d_out, int out_size, void* d_ws, size_t ws_size,
                              hipStream_t stream) {
    (void)in_sizes; (void)n_in; (void)out_size; (void)d_ws; (void)ws_size;
    const int*   x   = (const int*)  d_in[0];
    const float* emb = (const float*)d_in[1];
    const float* Wx  = (const float*)d_in[2];
    const float* Wh  = (const float*)d_in[3];
    const float* b   = (const float*)d_in[4];
    const float* Wd  = (const float*)d_in[5];
    const float* bd  = (const float*)d_in[6];
    float* out = (float*)d_out;

    dim3 grid(BATCH / 16);   // 256 waves, one 16-row batch tile each
    dim3 block(32);          // one wave32 per workgroup
    hipLaunchKernelGGL(rnn_wmma_kernel, grid, block, 0, stream,
                       x, emb, Wx, Wh, b, Wd, bd, out);
}
